// MIPTCellDemo_13529146982874
// MI455X (gfx1250) — compile-verified, matrix operand order checked by reading, not run on hardware
//
#include <hip/hip_runtime.h>
#include <stdint.h>

typedef __attribute__((ext_vector_type(16))) _Float16 v16h;
typedef __attribute__((ext_vector_type(8)))  _Float16 v8h;
typedef __attribute__((ext_vector_type(4)))  _Float16 v4h;
typedef __attribute__((ext_vector_type(8)))  float    v8f;
typedef __attribute__((ext_vector_type(4)))  int      v4i;

// AS-qualified 128-bit block types for the async-LDS builtin
typedef __attribute__((address_space(1))) v4i v4i_g;   // global
typedef __attribute__((address_space(3))) v4i v4i_l;   // LDS

constexpr int Bc = 4, Tc = 4096, Dc = 1024;
constexpr int Mc = Bc * Tc;        // 16384 rows
constexpr int Kc = Dc;             // 1024
constexpr int Nc = Dc;             // 1024
constexpr size_t BTD = (size_t)Bc * Tc * Dc;   // 16,777,216

// ---- CDNA5 async global->LDS copy + counter wait ---------------------------
__device__ __forceinline__ void async_copy_b128(uint32_t lds_off, const void* g) {
#if __has_builtin(__builtin_amdgcn_global_load_async_to_lds_b128)
  __builtin_amdgcn_global_load_async_to_lds_b128(
      (v4i_g*)g, (v4i_l*)(uintptr_t)lds_off, 0, 0);
#else
  asm volatile("global_load_async_to_lds_b128 %0, %1, off"
               :: "v"(lds_off), "v"(g) : "memory");
#endif
}

__device__ __forceinline__ void wait_async0() {
#if __has_builtin(__builtin_amdgcn_s_wait_asynccnt)
  __builtin_amdgcn_s_wait_asynccnt(0);
#else
  asm volatile("s_wait_asynccnt 0" ::: "memory");
#endif
}

// ---------------------------------------------------------------------------
// f32 -> f16 weight cast (1M elements per matrix)
// ---------------------------------------------------------------------------
__global__ void __launch_bounds__(256) cast_w(const float* __restrict__ w,
                                              _Float16* __restrict__ o) {
  size_t i = (size_t)blockIdx.x * blockDim.x + threadIdx.x;   // 4 elems each
  float4 v = ((const float4*)w)[i];
  v4h h;
  h.x = (_Float16)v.x; h.y = (_Float16)v.y;
  h.z = (_Float16)v.z; h.w = (_Float16)v.w;
  ((v4h*)o)[i] = h;
}

// ---------------------------------------------------------------------------
// Embedding gather + cast: one block per (b,t) row, 256 threads x 4 floats
// ---------------------------------------------------------------------------
__global__ void __launch_bounds__(256) gather_cast(const int* __restrict__ tok,
                                                   const float* __restrict__ emb,
                                                   _Float16* __restrict__ e16) {
  int row = blockIdx.x;                 // 0 .. Mc-1
  int t   = tok[row];
  float4 v = ((const float4*)(emb + (size_t)t * Dc))[threadIdx.x];
  v4h h;
  h.x = (_Float16)v.x; h.y = (_Float16)v.y;
  h.z = (_Float16)v.z; h.w = (_Float16)v.w;
  ((v4h*)(e16 + (size_t)row * Dc))[threadIdx.x] = h;
}

// ---------------------------------------------------------------------------
// WMMA GEMM with async-LDS double buffering.
//   out[m,n] = act( sum_k A[m,k] * W[n,k] + bias[n] )
// Block = 256 threads (8 waves).  Block tile: 128(M) x 64(N), K-step 32.
// Wave grid 4(M) x 2(N); each wave computes a 32x32 patch = 4 accumulators.
// LDS tile rows padded to 80B (20 dwords) -> conflict-free b128 frag reads.
// ---------------------------------------------------------------------------
constexpr int BM = 128, BN = 64, BK = 32;
constexpr int ROWB = 80;                         // padded row pitch (bytes)
constexpr int B_OFF = BM * ROWB;                 // B-tile base within buffer
constexpr int BUFB = (BM + BN) * ROWB;           // 15360 bytes per stage

__global__ void __launch_bounds__(256)
wmma_gemm(const _Float16* __restrict__ A, const _Float16* __restrict__ W,
          const float* __restrict__ bias, float* __restrict__ out, int sigm) {
  __shared__ __align__(16) char smem[2][BUFB];

  const int tid    = threadIdx.x;
  const int lane   = tid & 31;
  const int wave   = tid >> 5;
  const int wave_m = wave & 3;         // 0..3
  const int wave_n = wave >> 2;        // 0..1
  const int lr     = lane & 15;        // row-in-16 (A) / col-in-16 (B)
  const int hkB    = (lane >> 4) * 16; // half-wave K byte offset

  const int m0 = blockIdx.x * BM;
  const int n0 = blockIdx.y * BN;

  // --- async stage: 256 threads move A (8KB, 2 chunks) + B (4KB, 1 chunk) ---
  auto issue_tile = [&](int buf, int k) {
    uint32_t lbase = (uint32_t)(uintptr_t)&smem[buf][0];
#pragma unroll
    for (int c = 0; c < 2; ++c) {                 // A tile: 128 rows x 64B
      int off  = tid * 16 + c * 4096;
      int row  = off >> 6;
      int colB = off & 63;
      async_copy_b128(lbase + (uint32_t)(row * ROWB + colB),
                      A + (size_t)(m0 + row) * Kc + k + (colB >> 1));
    }
    {                                             // B tile: 64 rows x 64B
      int off  = tid * 16;
      int row  = off >> 6;
      int colB = off & 63;
      async_copy_b128(lbase + (uint32_t)(B_OFF + row * ROWB + colB),
                      W + (size_t)(n0 + row) * Kc + k + (colB >> 1));
    }
  };

  v8f acc[2][2];
#pragma unroll
  for (int i = 0; i < 2; ++i)
#pragma unroll
    for (int j = 0; j < 2; ++j)
      acc[i][j] = (v8f){0.f,0.f,0.f,0.f,0.f,0.f,0.f,0.f};

  issue_tile(0, 0);
  wait_async0();
  __syncthreads();

  int cur = 0;
  for (int k = 0; k < Kc; k += BK) {
    int nxt = cur ^ 1;
    if (k + BK < Kc) issue_tile(nxt, k + BK);     // overlap next-tile DMA

    const char* buf = smem[cur];
    v16h afr[2], bfr[2];
#pragma unroll
    for (int wm = 0; wm < 2; ++wm) {              // A fragments from LDS
      int off = (wave_m * 32 + wm * 16 + lr) * ROWB + hkB;
      v8h lo = *(const v8h*)(buf + off);
      v8h hi = *(const v8h*)(buf + off + 32);
      afr[wm] = __builtin_shufflevector(lo, hi, 0,1,2,3,4,5,6,7,
                                                8,9,10,11,12,13,14,15);
    }
#pragma unroll
    for (int wn = 0; wn < 2; ++wn) {              // B fragments from LDS
      int off = B_OFF + (wave_n * 32 + wn * 16 + lr) * ROWB + hkB;
      v8h lo = *(const v8h*)(buf + off);
      v8h hi = *(const v8h*)(buf + off + 32);
      bfr[wn] = __builtin_shufflevector(lo, hi, 0,1,2,3,4,5,6,7,
                                                8,9,10,11,12,13,14,15);
    }
#pragma unroll
    for (int wm = 0; wm < 2; ++wm)
#pragma unroll
      for (int wn = 0; wn < 2; ++wn)
        acc[wm][wn] = __builtin_amdgcn_wmma_f32_16x16x32_f16(
            false, afr[wm], false, bfr[wn], (short)0, acc[wm][wn],
            false, false);

    wait_async0();          // next tile landed in LDS
    __syncthreads();        // everyone done reading cur / writing nxt
    cur = nxt;
  }

  // C/D layout: VGPR r, lanes 0-15 -> M=r, N=lane; lanes 16-31 -> M=8+r
#pragma unroll
  for (int wm = 0; wm < 2; ++wm) {
    int rbase = m0 + wave_m * 32 + wm * 16 + (lane >> 4) * 8;
#pragma unroll
    for (int wn = 0; wn < 2; ++wn) {
      int col  = n0 + wave_n * 32 + wn * 16 + lr;
      float bv = bias[col];
#pragma unroll
      for (int r = 0; r < 8; ++r) {
        float v = acc[wm][wn][r] + bv;
        if (sigm) v = 1.0f / (1.0f + __expf(-v));
        out[(size_t)(rbase + r) * Nc + col] = v;
      }
    }
  }
}

// ---------------------------------------------------------------------------
// Sequential scan over T. One thread per (b,d) channel; coalesced across d.
// cos/sin folded in here (saves one full f32 stream of HBM traffic).
// ---------------------------------------------------------------------------
__global__ void __launch_bounds__(256)
scan_kernel(const float* __restrict__ theta, const float* __restrict__ ir,
            const float* __restrict__ ii, const float* __restrict__ p,
            float* __restrict__ hout) {
  int ch = blockIdx.x * blockDim.x + threadIdx.x;   // 0 .. B*D-1
  int b  = ch >> 10;                                // D = 1024
  int d  = ch & 1023;
  size_t base = (size_t)b * Tc * Dc + d;

  float hre = 0.f, him = 0.f;
  for (int t = 0; t < Tc; ++t) {
    size_t idx = base + (size_t)t * Dc;
    if (t + 8 < Tc) {
      size_t pf = idx + (size_t)8 * Dc;            // 8 steps ahead (4KB stride)
      __builtin_prefetch(&theta[pf], 0, 0);
      __builtin_prefetch(&p[pf], 0, 0);
      __builtin_prefetch(&ir[pf], 0, 0);
      __builtin_prefetch(&ii[pf], 0, 0);
    }
    float th = theta[idx];
    float pr = p[idx];
    float xr = ir[idx];
    float xi = ii[idx];
    float c = __cosf(th);
    float s = __sinf(th);
    float rre = c * hre - s * him;
    float rim = s * hre + c * him;
    hre = (1.0f - pr) * rre + pr * xr;
    him = (1.0f - pr) * rim + pr * xi;
    hout[idx] = hre;
  }
}

// ---------------------------------------------------------------------------
// Launch
// ---------------------------------------------------------------------------
extern "C" void kernel_launch(void* const* d_in, const int* in_sizes, int n_in,
                              void* d_out, int out_size, void* d_ws, size_t ws_size,
                              hipStream_t stream) {
  const int*   tokens = (const int*)d_in[0];
  const float* embed  = (const float*)d_in[1];
  const float* Ww[4]  = {(const float*)d_in[2], (const float*)d_in[4],
                         (const float*)d_in[6], (const float*)d_in[8]};
  const float* Wb[4]  = {(const float*)d_in[3], (const float*)d_in[5],
                         (const float*)d_in[7], (const float*)d_in[9]};
  float* out = (float*)d_out;

  // Workspace layout: e16 (32MiB) | w16 (8MiB) | theta | ir | ii (64MiB each)
  char* ws = (char*)d_ws;
  _Float16* e16 = (_Float16*)ws;
  _Float16* w16 = (_Float16*)(ws + (size_t)Mc * Kc * 2);
  float* theta  = (float*)(ws + (size_t)Mc * Kc * 2 + (size_t)4 * Nc * Kc * 2);
  float* irb    = theta + BTD;
  float* iib    = irb + BTD;

  float* h_out = out;          // output 0: h_all_re [B,T,D]
  float* p_out = out + BTD;    // output 1: p        [B,T,D]

  for (int z = 0; z < 4; ++z)
    cast_w<<<dim3((Nc * Kc / 4) / 256), dim3(256), 0, stream>>>(
        Ww[z], w16 + (size_t)z * Nc * Kc);

  gather_cast<<<dim3(Mc), dim3(256), 0, stream>>>(tokens, embed, e16);

  float* dsts[4] = {p_out, theta, irb, iib};
  for (int z = 0; z < 4; ++z)
    wmma_gemm<<<dim3(Mc / BM, Nc / BN), dim3(256), 0, stream>>>(
        e16, w16 + (size_t)z * Nc * Kc, Wb[z], dsts[z], z == 0 ? 1 : 0);

  scan_kernel<<<dim3((Bc * Dc) / 256), dim3(256), 0, stream>>>(
      theta, irb, iib, p_out, h_out);
}
